// GDM_1228360647124
// MI455X (gfx1250) — matrix operand order repaired and checked by reading, not compiled, not run on hardware
//
#include <hip/hip_runtime.h>
#include <hip/hip_bf16.h>

typedef __attribute__((ext_vector_type(16))) _Float16 v16h;
typedef __attribute__((ext_vector_type(8)))  float    v8f;

// ---------------- workspace layout (in 32-bit words) ----------------
static constexpr size_t OFF_POOLED = 0;         // [8,3,512,512] 6,291,456
static constexpr size_t OFF_IMGA   = 6291456;   // [8,512,512]   2,097,152
static constexpr size_t OFF_XGL    = 8388608;   // [8,512,512]   2,097,152
static constexpr size_t OFF_BUFA   = 10485760;  // [8,5,512,512] 10,485,760
static constexpr size_t OFF_BUFB   = 20971520;  // [8,5,512,512] 10,485,760
static constexpr size_t OFF_SMALL  = 31457280;
static constexpr int SMALL_WORDS   = 240 + 824; // floats + ints

// ---------------- init: zero accumulators ----------------
__global__ void init_small(unsigned* w) {
    int i = blockIdx.x * 256 + threadIdx.x;
    if (i < SMALL_WORDS) w[i] = 0u;
}

// ---------------- stage 1: 3x3 s2 avgpool + channel mean ----------------
__global__ __launch_bounds__(256) void pool3_kernel(const float* __restrict__ x,
                                                    float* __restrict__ pooled,
                                                    float* __restrict__ image_a) {
    int idx = blockIdx.x * 256 + threadIdx.x;          // 8*262144
    if (idx >= 8 * 262144) return;
    int b = idx >> 18, p = idx & 262143;
    int y = p >> 9, xq = p & 511;
    float csum = 0.f;
#pragma unroll
    for (int c = 0; c < 3; ++c) {
        const float* in = x + (size_t)(b * 3 + c) * 1048576;
        float s = 0.f;
#pragma unroll
        for (int dy = -1; dy <= 1; ++dy)
#pragma unroll
            for (int dx = -1; dx <= 1; ++dx) {
                int iy = 2 * y + dy, ix = 2 * xq + dx;
                if ((unsigned)iy < 1024u && (unsigned)ix < 1024u)
                    s += in[iy * 1024 + ix];
            }
        s *= (1.0f / 9.0f);
        pooled[((size_t)(b * 3 + c) * 512 + y) * 512 + xq] = s;
        csum += s;
    }
    image_a[idx] = csum * (1.0f / 3.0f);
}

// ---------------- stage 2: 7x7 s2 conv 3->5 as implicit GEMM via WMMA ----
// 8 waves/block, one 16-pixel tile per wave. K = 147 padded to 160 (5x32).
// kk,h are unroll-constants; only kgrp (lane>>4) is runtime, so each im2col
// offset is a cndmask-select between two compile-time constants.
__global__ __launch_bounds__(256) void conv_att_wmma(const float* __restrict__ pooled,
                                                     const float* __restrict__ conv_w,
                                                     float* __restrict__ acc) {
    const int wave = blockIdx.x * 8 + (threadIdx.x >> 5);  // 8*4096 tiles
    const int b    = wave >> 12;
    const int tile = wave & 4095;
    const int lane = threadIdx.x & 31;
    const int m    = lane & 15;
    const int kgrp = lane >> 4;           // 0 or 1

    const int n  = tile * 16 + m;         // output pixel (column of B)
    const int oy = n >> 8, ox = n & 255;
    const float* img = pooled + (size_t)b * 3 * 262144;

    // wave-uniform interior test: tile shares one oy; ox in [oxlo, oxlo+15]
    const int  oy0      = (tile * 16) >> 8;
    const int  oxlo     = (tile * 16) & 255;
    const bool interior = (oy0 >= 2) && (oy0 <= 253) && (oxlo >= 2) && (oxlo + 15 <= 253);

    v8f c = {};
    if (interior) {
        const float* base = img + (2 * oy - 3) * 512 + (2 * ox - 3);
#pragma unroll
        for (int kk = 0; kk < 5; ++kk) {
            v16h a;
#pragma unroll
            for (int h = 0; h < 16; ++h) {
                const int ka = kk * 32 + h + (h >= 8 ? 8 : 0);  // kgrp==0
                const int kb = ka + 8;                          // kgrp==1
                const int ia = (ka < 147) ? ka : 0;
                const int ib = (kb < 147) ? kb : 0;
                bool okk = kgrp ? (kb < 147) : (ka < 147);
                bool ok  = (m < 5) && okk;
                int  idx = m * 147 + (kgrp ? ib : ia);
                float w  = conv_w[ok ? idx : 0];
                a[h] = (_Float16)(ok ? w : 0.0f);
            }
            v16h bf;
#pragma unroll
            for (int h = 0; h < 16; ++h) {
                const int ka = kk * 32 + h + (h >= 8 ? 8 : 0);     // kgrp==0
                const int kb = ka + 8;         // kgrp==1
                const int oa = (ka < 147) ? ((ka / 49) * 262144 + ((ka % 49) / 7) * 512 + (ka % 7)) : 0;
                const int ob = (kb < 147) ? ((kb / 49) * 262144 + ((kb % 49) / 7) * 512 + (kb % 7)) : 0;
                bool okk = kgrp ? (kb < 147) : (ka < 147);
                int  off = kgrp ? ob : oa;      // select between two constants
                float v  = base[off];           // unconditional load
                bf[h] = (_Float16)(okk ? v : 0.0f);
            }
            c = __builtin_amdgcn_wmma_f32_16x16x32_f16(false, a, false, bf,
                                                       (short)0, c, false, false);
        }
    } else {
        // boundary tiles: branchless clamped loads
#pragma unroll
        for (int kk = 0; kk < 5; ++kk) {
            v16h a;
#pragma unroll
            for (int h = 0; h < 16; ++h) {
                const int ka = kk * 32 + h + (h >= 8 ? 8 : 0);
                const int kb = ka + 8;
                const int ia = (ka < 147) ? ka : 0;
                const int ib = (kb < 147) ? kb : 0;
                bool okk = kgrp ? (kb < 147) : (ka < 147);
                bool ok  = (m < 5) && okk;
                int  idx = m * 147 + (kgrp ? ib : ia);
                float w  = conv_w[ok ? idx : 0];
                a[h] = (_Float16)(ok ? w : 0.0f);
            }
            v16h bf;
#pragma unroll
            for (int h = 0; h < 16; ++h) {
                const int ka = kk * 32 + h + (h >= 8 ? 8 : 0);
                const int kb = ka + 8;
                const int kca = (ka < 147) ? ka : 0;
                const int kcb = (kb < 147) ? kb : 0;
                const int cca = kca / 49, kya = (kca % 49) / 7, kxa = kca % 7;
                const int ccb = kcb / 49, kyb = (kcb % 49) / 7, kxb = kcb % 7;
                bool okk = kgrp ? (kb < 147) : (ka < 147);
                int  cc = kgrp ? ccb : cca;
                int  ky = kgrp ? kyb : kya;
                int  kx = kgrp ? kxb : kxa;
                int  iy = 2 * oy - 3 + ky, ix = 2 * ox - 3 + kx;
                bool inb = okk && ((unsigned)iy < 512u) && ((unsigned)ix < 512u);
                int  iyc = min(max(iy, 0), 511);
                int  ixc = min(max(ix, 0), 511);
                float v = img[(size_t)cc * 262144 + iyc * 512 + ixc];
                bf[h] = (_Float16)(inb ? v : 0.0f);
            }
            c = __builtin_amdgcn_wmma_f32_16x16x32_f16(false, a, false, bf,
                                                       (short)0, c, false, false);
        }
    }

    // reduce each D row over its 16 columns (lanes 0-15: rows r, 16-31: rows r+8)
#pragma unroll
    for (int r = 0; r < 8; ++r) {
        float v = c[r];
        for (int off = 1; off < 16; off <<= 1) v += __shfl_xor(v, off, 32);
        c[r] = v;
    }
    if (lane == 0) {
#pragma unroll
        for (int r = 0; r < 5; ++r) atomicAdd(&acc[b * 5 + r], c[r]);
    }
}

// BN (affine, commutes with mean) + tanh -> weight_att[8,5]
__global__ void finish_att(const float* __restrict__ acc,
                           const float* __restrict__ g, const float* __restrict__ be,
                           const float* __restrict__ mu, const float* __restrict__ var,
                           float* __restrict__ watt) {
    int i = threadIdx.x;
    if (i < 40) {
        int oc = i % 5;
        float mval = acc[i] * (1.0f / 65536.0f);
        float v = (mval - mu[oc]) * (g[oc] * rsqrtf(var[oc] + 1e-5f)) + be[oc];
        watt[i] = tanhf(v);
    }
}

// ---------------- stage 3: GL conv + abs + stats/histogram ----------------
__global__ __launch_bounds__(256) void gl_kernel(const float* __restrict__ image_a,
                                                 const float* __restrict__ wparam,
                                                 float* __restrict__ xgl,
                                                 unsigned* __restrict__ maxbits,
                                                 int* __restrict__ all_sum,
                                                 int* __restrict__ hist) {
    int idx = blockIdx.x * 256 + threadIdx.x;   // 8*262144 (block spans one b)
    int b = idx >> 18, p = idx & 262143;
    int y = p >> 9, x = p & 511;
    float GL = wparam[0];
    float a1 = GL * 4.0f, a2 = -a1 * 0.125f, a3 = -a1 * 0.0625f;
    const float* img = image_a + (size_t)b * 262144;
    auto at = [&](int yy, int xx) -> float {
        return ((unsigned)yy < 512u && (unsigned)xx < 512u) ? img[yy * 512 + xx] : 0.f;
    };
    float s = a1 * at(y, x)
            + a2 * (at(y - 1, x) + at(y, x - 1) + at(y, x + 1) + at(y + 1, x))
            + a3 * (at(y - 2, x) + at(y - 1, x - 1) + at(y - 1, x + 1) + at(y, x - 2) +
                    at(y, x + 2) + at(y + 1, x - 1) + at(y + 1, x + 1) + at(y + 2, x));
    float v = fabsf(s);
    if (v < 1e-6f) v = 0.0f;
    xgl[idx] = v;

    __shared__ int sh[101];
    __shared__ int scount;
    for (int i = threadIdx.x; i < 101; i += 256) sh[i] = 0;
    if (threadIdx.x == 0) scount = 0;
    __syncthreads();
    if (v > 0.0f) {
        atomicAdd(&scount, 1);
        atomicMax(&maxbits[b], __float_as_uint(v));
        // smallest j with v <= xv[j], xv[j] = (float)(1e-5 * j^3)
        int j = (int)ceilf(cbrtf(v * 1e5f));
        if (j < 0) j = 0;
        if (j > 101) j = 101;
        while (j > 0) {
            float e = (float)(1e-5 * (double)(j - 1) * (j - 1) * (j - 1));
            if (v <= e) --j; else break;
        }
        while (j <= 100) {
            float e = (float)(1e-5 * (double)j * j * j);
            if (v > e) ++j; else break;
        }
        if (j <= 100) atomicAdd(&sh[j], 1);
    }
    __syncthreads();
    for (int i = threadIdx.x; i < 101; i += 256)
        if (sh[i]) atomicAdd(&hist[b * 101 + i], sh[i]);
    if (threadIdx.x == 0 && scount) atomicAdd(&all_sum[b], scount);
}

// per-batch: CDF -> 4 thresholds (first-argmin like jnp.argmin) -> band bounds
__global__ void compute_thresholds(const int* __restrict__ hist,
                                   const int* __restrict__ all_sum,
                                   const unsigned* __restrict__ maxbits,
                                   const float* __restrict__ watt,
                                   float* __restrict__ tmin, float* __restrict__ tmax) {
    int b = blockIdx.x;
    if (threadIdx.x != 0) return;
    float ps[4];
    for (int i = 0; i < 4; ++i) ps[i] = 0.2f + 0.2f * i + 0.05f * watt[b * 5 + (i + 1)];
    for (int i = 1; i < 4; ++i) {          // insertion sort (ascending)
        float v = ps[i]; int j = i - 1;
        while (j >= 0 && ps[j] > v) { ps[j + 1] = ps[j]; --j; }
        ps[j + 1] = v;
    }
    float inv = 1.0f / (float)all_sum[b];
    int best[4]; float bestd[4];
    for (int i = 0; i < 4; ++i) { best[i] = 0; bestd[i] = 3.4e38f; }
    int cum = 0;
    for (int j = 0; j < 101; ++j) {
        cum += hist[b * 101 + j];
        float pct = (float)cum * inv;
        for (int i = 0; i < 4; ++i) {
            float d = fabsf(pct - ps[i]);
            if (d < bestd[i]) { bestd[i] = d; best[i] = j; }
        }
    }
    float gmax = __uint_as_float(maxbits[b]);
    tmin[b * 5 + 0] = 0.0f;
    tmax[b * 5 + 4] = gmax;
    for (int i = 0; i < 4; ++i) {
        float t = (float)(1e-5 * (double)best[i] * best[i] * best[i]);
        tmin[b * 5 + i + 1] = t;
        tmax[b * 5 + i]     = t;
    }
}

// ---------------- stage 4: fused mask + horizontal 9-tap (pool1 H) ----------
__global__ __launch_bounds__(256) void maskpoolH(const float* __restrict__ xgl,
                                                 const float* __restrict__ tmin,
                                                 const float* __restrict__ tmax,
                                                 float* __restrict__ out) {
    int idx = blockIdx.x * 256 + threadIdx.x;   // 8*262144, all 5 channels per thread
    if (idx >= 8 * 262144) return;
    int b = idx >> 18, p = idx & 262143;
    int y = p >> 9, x = p & 511;
    float lo[5], hi[5];
#pragma unroll
    for (int c = 0; c < 5; ++c) { lo[c] = tmin[b * 5 + c]; hi[c] = tmax[b * 5 + c]; }
    const float* row = xgl + (size_t)b * 262144 + (size_t)y * 512;
    float acc[5] = {0, 0, 0, 0, 0};
#pragma unroll
    for (int dx = -4; dx <= 4; ++dx) {
        int xx = x + dx;
        if ((unsigned)xx < 512u) {
            float v = row[xx];
#pragma unroll
            for (int c = 0; c < 5; ++c)
                if (v > lo[c] && v <= hi[c]) acc[c] += 1.0f;
        }
    }
#pragma unroll
    for (int c = 0; c < 5; ++c)
        out[((size_t)(b * 5 + c) * 512 + y) * 512 + x] = acc[c];
}

// plain 9-tap horizontal pass on [8,5,512,512]
__global__ __launch_bounds__(256) void poolH(const float* __restrict__ in,
                                             float* __restrict__ out) {
    int idx = blockIdx.x * 256 + threadIdx.x;   // 8*5*262144
    if (idx >= 40 * 262144) return;
    int bc = idx >> 18, p = idx & 262143;
    int y = p >> 9, x = p & 511;
    const float* row = in + (size_t)bc * 262144 + (size_t)y * 512;
    float s = 0.f;
#pragma unroll
    for (int dx = -4; dx <= 4; ++dx) {
        int xx = x + dx;
        if ((unsigned)xx < 512u) s += row[xx];
    }
    out[idx] = s;
}

// 9-tap vertical pass with scale (1/81 completes each 9x9 avgpool)
__global__ __launch_bounds__(256) void poolV(const float* __restrict__ in,
                                             float* __restrict__ out, float scale) {
    int idx = blockIdx.x * 256 + threadIdx.x;   // 8*5*262144
    if (idx >= 40 * 262144) return;
    int bc = idx >> 18, p = idx & 262143;
    int y = p >> 9, x = p & 511;
    const float* img = in + (size_t)bc * 262144;
    float s = 0.f;
#pragma unroll
    for (int dy = -4; dy <= 4; ++dy) {
        int yy = y + dy;
        if ((unsigned)yy < 512u) s += img[yy * 512 + x];
    }
    out[idx] = s * scale;
}

// stride-2 horizontal: [8,5,512,512] -> [8,5,512,256]
__global__ __launch_bounds__(256) void poolHs2(const float* __restrict__ in,
                                               float* __restrict__ out) {
    int idx = blockIdx.x * 256 + threadIdx.x;   // 8*5*512*256
    if (idx >= 40 * 131072) return;
    int bc = idx >> 17, p = idx & 131071;
    int y = p >> 8, x2 = p & 255;
    const float* row = in + (size_t)bc * 262144 + (size_t)y * 512;
    float s = 0.f;
#pragma unroll
    for (int dx = -4; dx <= 4; ++dx) {
        int xx = 2 * x2 + dx;
        if ((unsigned)xx < 512u) s += row[xx];
    }
    out[idx] = s;
}

// stride-2 vertical: [8,5,512,256] -> [8,5,256,256], scaled
__global__ __launch_bounds__(256) void poolVs2(const float* __restrict__ in,
                                               float* __restrict__ out, float scale) {
    int idx = blockIdx.x * 256 + threadIdx.x;   // 8*5*256*256
    if (idx >= 40 * 65536) return;
    int bc = idx >> 16, p = idx & 65535;
    int y2 = p >> 8, x2 = p & 255;
    const float* img = in + (size_t)bc * 131072;
    float s = 0.f;
#pragma unroll
    for (int dy = -4; dy <= 4; ++dy) {
        int yy = 2 * y2 + dy;
        if ((unsigned)yy < 512u) s += img[yy * 256 + x2];
    }
    out[idx] = s * scale;
}

// per-(b,c) min/max over 65536 values
__global__ __launch_bounds__(256) void minmax_kernel(const float* __restrict__ in,
                                                     float* __restrict__ mn,
                                                     float* __restrict__ mx) {
    int bc = blockIdx.x;
    const float* p = in + (size_t)bc * 65536;
    float lmn = 3.4e38f, lmx = -3.4e38f;
    for (int i = threadIdx.x; i < 65536; i += 256) {
        float v = p[i];
        lmn = fminf(lmn, v);
        lmx = fmaxf(lmx, v);
    }
    __shared__ float smn[256], smx[256];
    smn[threadIdx.x] = lmn; smx[threadIdx.x] = lmx;
    __syncthreads();
    for (int s = 128; s > 0; s >>= 1) {
        if (threadIdx.x < s) {
            smn[threadIdx.x] = fminf(smn[threadIdx.x], smn[threadIdx.x + s]);
            smx[threadIdx.x] = fmaxf(smx[threadIdx.x], smx[threadIdx.x + s]);
        }
        __syncthreads();
    }
    if (threadIdx.x == 0) { mn[bc] = smn[0]; mx[bc] = smx[0]; }
}

__global__ __launch_bounds__(256) void normalize_kernel(float* __restrict__ out,
                                                        const float* __restrict__ mn,
                                                        const float* __restrict__ mx) {
    int idx = blockIdx.x * 256 + threadIdx.x;
    if (idx >= 40 * 65536) return;
    int bc = idx >> 16;
    float a = mn[bc];
    out[idx] = (out[idx] - a) / (mx[bc] - a);
}

extern "C" void kernel_launch(void* const* d_in, const int* in_sizes, int n_in,
                              void* d_out, int out_size, void* d_ws, size_t ws_size,
                              hipStream_t stream) {
    const float* x       = (const float*)d_in[0];   // [8,3,1024,1024]
    const float* weight  = (const float*)d_in[1];   // [1,5]
    const float* conv_w  = (const float*)d_in[2];   // [5,3,7,7]
    const float* bn_g    = (const float*)d_in[3];
    const float* bn_b    = (const float*)d_in[4];
    const float* bn_m    = (const float*)d_in[5];
    const float* bn_v    = (const float*)d_in[6];
    float* out = (float*)d_out;                     // [8,5,256,256]

    float* ws      = (float*)d_ws;
    float* pooled  = ws + OFF_POOLED;
    float* image_a = ws + OFF_IMGA;
    float* xgl     = ws + OFF_XGL;
    float* bufA    = ws + OFF_BUFA;
    float* bufB    = ws + OFF_BUFB;
    float* acc     = ws + OFF_SMALL;        // 40
    float* watt    = acc + 40;              // 40
    float* tmin    = watt + 40;             // 40
    float* tmax    = tmin + 40;             // 40
    float* mn      = tmax + 40;             // 40
    float* mx      = mn + 40;               // 40
    int*      hist    = (int*)(ws + OFF_SMALL + 240);   // 8*101
    int*      all_sum = hist + 808;                     // 8
    unsigned* maxbits = (unsigned*)(all_sum + 8);       // 8

    const float inv81 = 1.0f / 81.0f;

    init_small<<<(SMALL_WORDS + 255) / 256, 256, 0, stream>>>((unsigned*)(ws + OFF_SMALL));

    pool3_kernel<<<8192, 256, 0, stream>>>(x, pooled, image_a);

    conv_att_wmma<<<4096, 256, 0, stream>>>(pooled, conv_w, acc);
    finish_att<<<1, 64, 0, stream>>>(acc, bn_g, bn_b, bn_m, bn_v, watt);

    gl_kernel<<<8192, 256, 0, stream>>>(image_a, weight, xgl, maxbits, all_sum, hist);
    compute_thresholds<<<8, 32, 0, stream>>>(hist, all_sum, maxbits, watt, tmin, tmax);

    // pool1 (9x9 s1) fused with band-mask generation
    maskpoolH<<<8192, 256, 0, stream>>>(xgl, tmin, tmax, bufA);
    poolV<<<40960, 256, 0, stream>>>(bufA, bufB, inv81);
    // pool2 (9x9 s1)
    poolH<<<40960, 256, 0, stream>>>(bufB, bufA);
    poolV<<<40960, 256, 0, stream>>>(bufA, bufB, inv81);
    // pool3 (9x9 s2) -> d_out
    poolHs2<<<20480, 256, 0, stream>>>(bufB, bufA);
    poolVs2<<<10240, 256, 0, stream>>>(bufA, out, inv81);

    minmax_kernel<<<40, 256, 0, stream>>>(out, mn, mx);
    normalize_kernel<<<10240, 256, 0, stream>>>(out, mn, mx);
}
